// ResidualStack_77369540870492
// MI455X (gfx1250) — matrix operand order repaired and checked by reading, not compiled, not run on hardware
//
#include <hip/hip_runtime.h>

#define T_LEN     16384
#define BATCH     4
#define RES_CH    128
#define SKIP_CH   256
#define N_BLOCKS  40
#define LAYER_SZ  10
#define TT        64          // timesteps per workgroup tile
#define NT        (TT / 16)   // 16x16 tiles along time = 4
#define XS        264         // LDS row stride (halfs) for X tile: 256 cols + 8 pad (16B align kept)
#define GS        136         // LDS row stride (halfs) for gated tile: 128 cols + 8 pad

typedef __attribute__((ext_vector_type(16))) __bf16         v16bf;
typedef __attribute__((ext_vector_type(2)))  __bf16         v2bf;
typedef __attribute__((ext_vector_type(8)))  float          v8f;
typedef __attribute__((ext_vector_type(8)))  unsigned short ushort8;

union Frag16 { unsigned short s[16]; ushort8 u8[2]; v16bf v; };

// Native bf16 conversion: hardware cvt when available, RNE software seq otherwise.
__device__ __forceinline__ unsigned short f2bf(float f) {
  __bf16 h = (__bf16)f;
  return __builtin_bit_cast(unsigned short, h);
}

// Packed pair conversion -> one u32
__device__ __forceinline__ unsigned int f2bf_pk(float a, float b) {
  v2bf r;
  r.x = (__bf16)a;
  r.y = (__bf16)b;
  return __builtin_bit_cast(unsigned int, r);
}

// 16 K-contiguous bf16 from LDS -> B fragment (two 16B chunks)
__device__ __forceinline__ v16bf frag_b16(const unsigned short* base) {
  Frag16 f;
  f.u8[0] = *(const ushort8*)(base);
  f.u8[1] = *(const ushort8*)(base + 8);
  return f.v;
}

// A fragment from pre-converted bf16 weight row: two global 16B loads
__device__ __forceinline__ v16bf frag_a_g(const unsigned short* __restrict__ row,
                                          int kb, int lhalf) {
  Frag16 f;
  f.u8[0] = *(const ushort8*)(row + kb + lhalf * 16);
  f.u8[1] = *(const ushort8*)(row + kb + lhalf * 16 + 8);
  return f.v;
}

// ---------------- one-time weight pre-conversion (f32 -> bf16 caches) ----------------
__global__ __launch_bounds__(256) void convert_weights_kernel(
    const float* __restrict__ wdil, const float* __restrict__ wres,
    const float* __restrict__ wskp, unsigned short* __restrict__ wdB,
    unsigned short* __restrict__ wrB, unsigned short* __restrict__ wsB)
{
  const size_t tid    = (size_t)blockIdx.x * blockDim.x + threadIdx.x;
  const size_t stride = (size_t)gridDim.x * blockDim.x;

  // dilated conv: [N*128][256] bf16, K = [tap1 c0..127 | tap0 c0..127]
  const size_t ndil = (size_t)N_BLOCKS * RES_CH * 256;
  for (size_t idx = tid; idx < ndil; idx += stride) {
    size_t io = idx >> 8;               // block*128 + o
    int k = (int)(idx & 255);
    float v = (k < RES_CH) ? wdil[(io * RES_CH + k) * 2 + 1]
                           : wdil[(io * RES_CH + (k - RES_CH)) * 2 + 0];
    wdB[idx] = f2bf(v);
  }
  const size_t nres = (size_t)N_BLOCKS * RES_CH * RES_CH;
  for (size_t idx = tid; idx < nres; idx += stride) wrB[idx] = f2bf(wres[idx]);
  const size_t nskp = (size_t)N_BLOCKS * SKIP_CH * RES_CH;
  for (size_t idx = tid; idx < nskp; idx += stride) wsB[idx] = f2bf(wskp[idx]);
}

// ---------------- one residual block ----------------
__global__ __launch_bounds__(256) void wavenet_block_kernel(
    const float* __restrict__ xin, float* __restrict__ xout,
    const unsigned short* __restrict__ wd, const unsigned short* __restrict__ wr,
    const unsigned short* __restrict__ wsk, float* __restrict__ skip_out,
    int dilation, int skip_size)
{
  __shared__ unsigned short sX[TT * XS];   // [t][c], c: 0..127 = X[t], 128..255 = X[t-d]
  __shared__ unsigned short sG[TT * GS];   // [t][c] gated activations (bf16)

  const int tilesPerB = T_LEN / TT;
  const int b   = blockIdx.x / tilesPerB;
  const int t0  = (blockIdx.x % tilesPerB) * TT;

  const int tid   = threadIdx.x;
  const int wv    = tid >> 5;          // wave id 0..7
  const int lane  = tid & 31;
  const int lmod  = lane & 15;
  const int lhalf = lane >> 4;         // 0 | 1

  const int tskip0 = T_LEN - skip_size;
  const bool doSkip = (t0 >= tskip0);

  // ---- stage X tile (current + dilated past tap) into LDS, packed-pair bf16 ----
  for (int idx = tid; idx < (SKIP_CH / 2) * TT; idx += 256) {
    int cp = idx >> 6;                 // channel-pair 0..127 (TT == 64)
    int t  = idx & (TT - 1);
    int c  = cp * 2;
    float v0, v1;
    if (c < RES_CH) {
      const float* p = &xin[(size_t)(b * RES_CH + c) * T_LEN + t0 + t];
      v0 = p[0];
      v1 = p[T_LEN];
    } else {
      int tp = t0 + t - dilation;
      const float* p = &xin[(size_t)(b * RES_CH + (c - RES_CH)) * T_LEN + tp];
      v0 = (tp >= 0) ? p[0]     : 0.0f;
      v1 = (tp >= 0) ? p[T_LEN] : 0.0f;
    }
    *(unsigned int*)&sX[t * XS + c] = f2bf_pk(v0, v1);
  }

  // ---- A fragments for dilated conv (pre-concatenated bf16 rows, K=256) ----
  v16bf afr[8];
  {
    const unsigned short* row = wd + (size_t)((wv << 4) + lmod) * 256;
    #pragma unroll
    for (int kk = 0; kk < 8; ++kk) afr[kk] = frag_a_g(row, kk * 32, lhalf);
  }

  // ---- residual C-matrix (f32) loaded early: latency hides under stage 1 ----
  v8f racc[NT];
  #pragma unroll
  for (int nt = 0; nt < NT; ++nt) {
    int t = nt * 16 + lmod;
    #pragma unroll
    for (int r = 0; r < 8; ++r) {
      int o = (wv << 4) + r + (lhalf << 3);
      racc[nt][r] = xin[(size_t)(b * RES_CH + o) * T_LEN + t0 + t];
    }
  }

  __syncthreads();

  // ---- Stage 1: h = Wdil x [Xcur;Xprev], B double-buffered over kk ----
  const unsigned short* brows[NT];
  #pragma unroll
  for (int nt = 0; nt < NT; ++nt)
    brows[nt] = &sX[(nt * 16 + lmod) * XS + lhalf * 16];

  v8f hacc[NT] = {};
  {
    v16bf bcur[NT], bnxt[NT];
    #pragma unroll
    for (int nt = 0; nt < NT; ++nt) bcur[nt] = frag_b16(brows[nt]);
    #pragma unroll
    for (int kk = 0; kk < 8; ++kk) {
      #pragma unroll
      for (int nt = 0; nt < NT; ++nt)
        if (kk < 7) bnxt[nt] = frag_b16(brows[nt] + (kk + 1) * 32);
      #pragma unroll
      for (int nt = 0; nt < NT; ++nt)
        hacc[nt] = __builtin_amdgcn_wmma_f32_16x16x32_bf16(false, afr[kk], false, bcur[nt],
                                                           (short)0, hacc[nt], false, false);
      #pragma unroll
      for (int nt = 0; nt < NT; ++nt)
        if (kk < 7) bcur[nt] = bnxt[nt];
    }
  }

  // ---- gated = tanh(h)*sigmoid(h) -> sG (packed-pair bf16 stores) ----
  #pragma unroll
  for (int nt = 0; nt < NT; ++nt) {
    int t = nt * 16 + lmod;
    #pragma unroll
    for (int r = 0; r < 8; r += 2) {
      float g[2];
      #pragma unroll
      for (int q = 0; q < 2; ++q) {
        float h   = hacc[nt][r + q];
        float e   = __expf(-h);
        float sig = __fdividef(1.0f, 1.0f + e);
        float e2  = e * e;
        float th  = __fdividef(1.0f - e2, 1.0f + e2);   // tanh(h)
        g[q] = th * sig;
      }
      int o = (wv << 4) + r + (lhalf << 3);
      *(unsigned int*)&sG[t * GS + o] = f2bf_pk(g[0], g[1]);
    }
  }

  // ---- A fragments for residual + skip 1x1 (global bf16, overlap barrier) ----
  v16bf ars[4];
  {
    const unsigned short* row = wr + (size_t)((wv << 4) + lmod) * RES_CH;
    #pragma unroll
    for (int kk = 0; kk < 4; ++kk) ars[kk] = frag_a_g(row, kk * 32, lhalf);
  }
  v16bf ask[2][4];
  if (doSkip) {
    #pragma unroll
    for (int half = 0; half < 2; ++half) {
      const int ob = (wv + half * 8) << 4;
      const unsigned short* row = wsk + (size_t)(ob + lmod) * RES_CH;
      #pragma unroll
      for (int kk = 0; kk < 4; ++kk) ask[half][kk] = frag_a_g(row, kk * 32, lhalf);
    }
  }

  __syncthreads();   // sG fully written

  const unsigned short* grows[NT];
  #pragma unroll
  for (int nt = 0; nt < NT; ++nt)
    grows[nt] = &sG[(nt * 16 + lmod) * GS + lhalf * 16];

  // ---- Stage 2a: out = Wres x gated + xin (residual pre-loaded in racc) ----
  {
    v16bf gcur[NT], gnxt[NT];
    #pragma unroll
    for (int nt = 0; nt < NT; ++nt) gcur[nt] = frag_b16(grows[nt]);
    #pragma unroll
    for (int kk = 0; kk < 4; ++kk) {
      #pragma unroll
      for (int nt = 0; nt < NT; ++nt)
        if (kk < 3) gnxt[nt] = frag_b16(grows[nt] + (kk + 1) * 32);
      #pragma unroll
      for (int nt = 0; nt < NT; ++nt)
        racc[nt] = __builtin_amdgcn_wmma_f32_16x16x32_bf16(false, ars[kk], false, gcur[nt],
                                                           (short)0, racc[nt], false, false);
      #pragma unroll
      for (int nt = 0; nt < NT; ++nt)
        if (kk < 3) gcur[nt] = gnxt[nt];
    }
  }
  #pragma unroll
  for (int nt = 0; nt < NT; ++nt) {
    int t = nt * 16 + lmod;
    #pragma unroll
    for (int r = 0; r < 8; ++r) {
      int o = (wv << 4) + r + (lhalf << 3);
      xout[(size_t)(b * RES_CH + o) * T_LEN + t0 + t] = racc[nt][r];
    }
  }

  // ---- Stage 2b: skip = Wskip x gated; both 128-row halves share B frags ----
  if (doSkip) {
    v8f sacc[2][NT] = {};
    {
      v16bf gcur[NT], gnxt[NT];
      #pragma unroll
      for (int nt = 0; nt < NT; ++nt) gcur[nt] = frag_b16(grows[nt]);
      #pragma unroll
      for (int kk = 0; kk < 4; ++kk) {
        #pragma unroll
        for (int nt = 0; nt < NT; ++nt)
          if (kk < 3) gnxt[nt] = frag_b16(grows[nt] + (kk + 1) * 32);
        #pragma unroll
        for (int half = 0; half < 2; ++half)
          #pragma unroll
          for (int nt = 0; nt < NT; ++nt)
            sacc[half][nt] = __builtin_amdgcn_wmma_f32_16x16x32_bf16(
                false, ask[half][kk], false, gcur[nt], (short)0, sacc[half][nt], false, false);
        #pragma unroll
        for (int nt = 0; nt < NT; ++nt)
          if (kk < 3) gcur[nt] = gnxt[nt];
      }
    }
    #pragma unroll
    for (int half = 0; half < 2; ++half) {
      const int ob = (wv + half * 8) << 4;
      #pragma unroll
      for (int nt = 0; nt < NT; ++nt) {
        int t = nt * 16 + lmod;
        #pragma unroll
        for (int r = 0; r < 8; ++r) {
          int o = ob + r + (lhalf << 3);
          skip_out[(size_t)(b * SKIP_CH + o) * skip_size + (t0 + t - tskip0)] = sacc[half][nt][r];
        }
      }
    }
  }
}

extern "C" void kernel_launch(void* const* d_in, const int* in_sizes, int n_in,
                              void* d_out, int out_size, void* d_ws, size_t ws_size,
                              hipStream_t stream) {
  const float* x    = (const float*)d_in[0];
  const float* wdil = (const float*)d_in[1];
  const float* wres = (const float*)d_in[2];
  const float* wskp = (const float*)d_in[3];
  float* out = (float*)d_out;

  // skip_size recoverable host-side from output geometry (no D2H copy needed)
  const int skip_size = out_size / (N_BLOCKS * BATCH * SKIP_CH);

  // workspace layout: [buf0 f32][buf1 f32][wd bf16][wr bf16][ws bf16]
  const size_t actElems = (size_t)BATCH * RES_CH * T_LEN;
  float* buf0 = (float*)d_ws;
  float* buf1 = buf0 + actElems;
  unsigned short* wdB = (unsigned short*)(buf1 + actElems);
  unsigned short* wrB = wdB + (size_t)N_BLOCKS * RES_CH * 256;
  unsigned short* wsB = wrB + (size_t)N_BLOCKS * RES_CH * RES_CH;

  // one-time weight conversion (runs every call; deterministic, graph-safe)
  convert_weights_kernel<<<dim3(1024), dim3(256), 0, stream>>>(wdil, wres, wskp,
                                                               wdB, wrB, wsB);

  dim3 grid(BATCH * (T_LEN / TT));
  dim3 block(256);

  for (int i = 0; i < N_BLOCKS; ++i) {
    int d = 1 << (i % LAYER_SZ);
    const float* xin = (i == 0) ? x : ((i & 1) ? buf0 : buf1);
    float*      xout = (i & 1) ? buf1 : buf0;
    const unsigned short* wd = wdB + (size_t)i * RES_CH * 256;
    const unsigned short* wr = wrB + (size_t)i * RES_CH * RES_CH;
    const unsigned short* ws = wsB + (size_t)i * SKIP_CH * RES_CH;
    float* so = out + (size_t)i * BATCH * SKIP_CH * skip_size;
    wavenet_block_kernel<<<grid, block, 0, stream>>>(xin, xout, wd, wr, ws, so, d, skip_size);
  }
}
